// SproutGNN_17514876634166
// MI455X (gfx1250) — compile-verified, matrix operand order checked by reading, not run on hardware
//
#include <hip/hip_runtime.h>
#include <hip/hip_bf16.h>
#include <math.h>

// ---------------------------------------------------------------------------
// SproutGNN forward for MI455X (gfx1250, wave32, WMMA)
// ---------------------------------------------------------------------------

typedef __bf16 bf16;
typedef __attribute__((ext_vector_type(16))) __bf16 bf16x16;
typedef __attribute__((ext_vector_type(8)))  float  f32x8;

#define NN     4096
#define INC    512
#define HIDC   512
#define OUTC   40
#define OUTP   48
#define KCL    7
#define PK     10
#define PKP    16
#define COMBC  2048
#define WORDS  128     // 4096 bits / 32

#define EPI_BIAS     1
#define EPI_ROWSCALE 4
#define EPI_BF16OUT  8

// ---------------------------------------------------------------------------
// WMMA bf16 GEMM:  C[M,N] = A[M,K] @ Bt[N,K]^T   (+bias, *rowscale, bf16out)
// Both operands row-major with contiguous K -> all fragment loads are b128.
// Block: 256 threads = 8 waves (4 along M x 2 along N); wave tile 32x64
// (2x4 f32x8 accumulators, 8 WMMAs per K-step). Block tile 128x128.
// FLAGS is a template parameter -> straight-line epilogue (no exec branches).
// ---------------------------------------------------------------------------
__device__ __forceinline__ bf16x16 load_frag(const bf16* __restrict__ base, int hb)
{
    union { unsigned u[8]; bf16x16 v; } f;
#pragma unroll
    for (int r = 0; r < 8; ++r) {
        int kb = ((r >> 2) << 4) + (hb << 3) + ((r & 3) << 1);
        f.u[r] = *(const unsigned*)(base + kb);   // 2 x bf16 per dword
    }
    return f.v;
}

template <int FLAGS>
__global__ __launch_bounds__(256)
void k_wmma_gemm_bf16(const bf16* __restrict__ A, const bf16* __restrict__ Bt,
                      const float* __restrict__ bias,
                      const float* __restrict__ rowscale,
                      void* __restrict__ C, int M, int N, int K)
{
    const int lane = threadIdx.x & 31;
    const int wave = threadIdx.x >> 5;                  // 0..7
    const int m0 = blockIdx.y * 128 + (wave & 3) * 32;  // 4 waves along M
    const int n0 = blockIdx.x * 128 + (wave >> 2) * 64; // 2 waves along N
    const int mrow = lane & 15;
    const int hb   = lane >> 4;

    if (m0 >= M) return;

    f32x8 acc[2][4] = {{{}, {}, {}, {}}, {{}, {}, {}, {}}};

    // clamp OOB fragment rows to a safe row (never stored)
    bool nok[4];
    size_t brow[4];
#pragma unroll
    for (int j = 0; j < 4; ++j) {
        int nt = n0 + 16 * j;
        nok[j]  = (nt + 16) <= N;
        brow[j] = (size_t)(nok[j] ? nt + mrow : mrow) * K;
    }
    const size_t arow0 = (size_t)(m0 + mrow) * K;
    const size_t arow1 = (size_t)(m0 + 16 + mrow) * K;

    for (int kk = 0; kk < K; kk += 32) {
        bf16x16 a0 = load_frag(A + arow0 + kk, hb);
        bf16x16 a1 = load_frag(A + arow1 + kk, hb);
        __builtin_prefetch(A + arow0 + kk + 64, 0, 1);
        __builtin_prefetch(A + arow1 + kk + 64, 0, 1);
        bf16x16 b[4];
#pragma unroll
        for (int j = 0; j < 4; ++j) b[j] = load_frag(Bt + brow[j] + kk, hb);
#pragma unroll
        for (int j = 0; j < 4; ++j) {
            acc[0][j] = __builtin_amdgcn_wmma_f32_16x16x32_bf16(
                false, a0, false, b[j], (short)0, acc[0][j], false, false);
            acc[1][j] = __builtin_amdgcn_wmma_f32_16x16x32_bf16(
                false, a1, false, b[j], (short)0, acc[1][j], false, false);
        }
    }

    // ---- epilogue: ISA f32 C/D layout (VGPR r -> M = r + 8*half) ----
#pragma unroll
    for (int j = 0; j < 4; ++j) {
        if (!nok[j]) continue;
        const int n = n0 + 16 * j + mrow;
        const float bn = (FLAGS & EPI_BIAS) ? bias[n] : 0.f;
#pragma unroll
        for (int i = 0; i < 2; ++i) {
            const int mb = m0 + 16 * i + (hb << 3);
#pragma unroll
            for (int r = 0; r < 8; ++r) {
                float v = acc[i][j][r];
                if (FLAGS & EPI_BIAS)     v += bn;
                if (FLAGS & EPI_ROWSCALE) v *= rowscale[mb + r];
                if (FLAGS & EPI_BF16OUT)
                    ((bf16*)C)[(size_t)(mb + r) * N + n] = (bf16)v;
                else
                    ((float*)C)[(size_t)(mb + r) * N + n] = v;
            }
        }
    }
}

// ---------------------------------------------------------------------------
// Ego branch: exact 2-hop reachability via bitsets
// ---------------------------------------------------------------------------
__global__ void k_adjbits(const int* __restrict__ src, const int* __restrict__ dst,
                          unsigned* __restrict__ adj, int E)
{
    int e = blockIdx.x * blockDim.x + threadIdx.x;
    if (e >= E) return;
    int s = src[e], d = dst[e];
    atomicOr(&adj[(size_t)d * WORDS + (s >> 5)], 1u << (s & 31));
}

__global__ void k_reach_init(const unsigned* __restrict__ adj, unsigned* __restrict__ reach)
{
    int idx = blockIdx.x * blockDim.x + threadIdx.x;
    if (idx >= NN * WORDS) return;
    int i = idx >> 7, w = idx & (WORDS - 1);
    unsigned v = adj[idx];
    if (w == (i >> 5)) v |= 1u << (i & 31);   // self
    reach[idx] = v;
}

__global__ __launch_bounds__(128)
void k_reach_2hop(const int* __restrict__ src, const int* __restrict__ dst,
                  const unsigned* __restrict__ adj, unsigned* __restrict__ reach)
{
    int e = blockIdx.x, w = threadIdx.x;
    int s = src[e], d = dst[e];
    unsigned bits = adj[(size_t)s * WORDS + w];
    if (bits) atomicOr(&reach[(size_t)d * WORDS + w], bits);
}

__global__ void k_mf_expand(const unsigned* __restrict__ reach, bf16* __restrict__ Mf)
{
    int idx = blockIdx.x * blockDim.x + threadIdx.x;
    if (idx >= NN * WORDS) return;
    int i = idx >> 7, w = idx & (WORDS - 1);
    unsigned bits = reach[idx];
    unsigned words[16];
#pragma unroll
    for (int b = 0; b < 16; ++b) {
        unsigned lo = ((bits >> (2 * b)) & 1u)     ? 0x00003F80u : 0u;   // bf16 1.0
        unsigned hi = ((bits >> (2 * b + 1)) & 1u) ? 0x3F800000u : 0u;
        words[b] = lo | hi;
    }
    uint4* row = (uint4*)(Mf + (size_t)i * NN + w * 32);
#pragma unroll
    for (int q = 0; q < 4; ++q)
        row[q] = make_uint4(words[4 * q], words[4 * q + 1],
                            words[4 * q + 2], words[4 * q + 3]);
}

__global__ void k_rowrecip(const unsigned* __restrict__ reach, float* __restrict__ rr)
{
    int i = blockIdx.x * blockDim.x + threadIdx.x;
    if (i >= NN) return;
    int cnt = 0;
    for (int w = 0; w < WORDS; ++w) cnt += __popc(reach[(size_t)i * WORDS + w]);
    rr[i] = 1.f / (float)(cnt < 1 ? 1 : cnt);
}

// ---------------------------------------------------------------------------
// Conversions / prep
// ---------------------------------------------------------------------------
__global__ void k_prep_x(const float* __restrict__ x, bf16* __restrict__ xb,
                         bf16* __restrict__ xbT)
{
    int idx = blockIdx.x * blockDim.x + threadIdx.x;
    if (idx >= NN * INC) return;
    int i = idx >> 9, c = idx & (INC - 1);
    float v = x[idx];
    xb[idx] = (bf16)v;
    xbT[(size_t)c * NN + i] = (bf16)v;
}

__global__ void k_cvt_bf16_t(const float* __restrict__ W, bf16* __restrict__ Wt,
                             int rows, int cols)
{
    int idx = blockIdx.x * blockDim.x + threadIdx.x;
    if (idx >= rows * cols) return;
    int r = idx / cols, c = idx % cols;
    Wt[(size_t)c * rows + r] = (bf16)W[idx];
}

__global__ __launch_bounds__(256)
void k_colmean(const float* __restrict__ x, float* __restrict__ mean)
{
    int c = blockIdx.x;
    __shared__ float red[256];
    float s = 0.f;
    for (int i = threadIdx.x; i < NN; i += 256) s += x[(size_t)i * INC + c];
    red[threadIdx.x] = s; __syncthreads();
    for (int s2 = 128; s2 > 0; s2 >>= 1) {
        if (threadIdx.x < s2) red[threadIdx.x] += red[threadIdx.x + s2];
        __syncthreads();
    }
    if (threadIdx.x == 0) mean[c] = red[0] / (float)NN;
}

__global__ void k_center(const float* __restrict__ x, const float* __restrict__ mean,
                         float* __restrict__ xc, bf16* __restrict__ xcTb)
{
    int idx = blockIdx.x * blockDim.x + threadIdx.x;
    if (idx >= NN * INC) return;
    int i = idx >> 9, c = idx & (INC - 1);
    float v = x[idx] - mean[c];
    xc[idx] = v;
    xcTb[(size_t)c * NN + i] = (bf16)v;
}

// ---------------------------------------------------------------------------
// Power iteration (top-10 subspace of covariance) + Gram-Schmidt
// ---------------------------------------------------------------------------
__global__ void k_init_V(float* __restrict__ V)
{
    int idx = blockIdx.x * blockDim.x + threadIdx.x;
    if (idx >= INC * PKP) return;
    int r = idx >> 4, j = idx & 15;
    if (j >= PK) { V[idx] = 0.f; return; }
    unsigned h = (unsigned)r * 1664525u + (unsigned)j * 1013904223u + 12345u;
    h ^= h >> 13; h *= 0x5bd1e995u; h ^= h >> 15;
    V[idx] = ((float)(h & 0xFFFFu) / 65536.f) - 0.5f;
}

__global__ void k_cov_apply(const float* __restrict__ Cm, const float* __restrict__ V,
                            float* __restrict__ U)
{
    int r = blockIdx.x * blockDim.x + threadIdx.x;
    if (r >= INC) return;
    float acc[PK] = {};
    for (int k = 0; k < INC; ++k) {
        float c = Cm[(size_t)r * INC + k];
#pragma unroll
        for (int j = 0; j < PK; ++j) acc[j] += c * V[k * PKP + j];
    }
#pragma unroll
    for (int j = 0; j < PK; ++j) U[r * PKP + j] = acc[j];
}

__global__ __launch_bounds__(512)
void k_gram_schmidt(const float* __restrict__ U, float* __restrict__ V)
{
    __shared__ float col[PK][INC];
    __shared__ float red[INC];
    int t = threadIdx.x;
    for (int j = 0; j < PK; ++j) col[j][t] = U[t * PKP + j];
    __syncthreads();
    for (int j = 0; j < PK; ++j) {
        for (int i = 0; i < j; ++i) {
            red[t] = col[j][t] * col[i][t]; __syncthreads();
            for (int s2 = 256; s2 > 0; s2 >>= 1) {
                if (t < s2) red[t] += red[t + s2];
                __syncthreads();
            }
            float d = red[0]; __syncthreads();
            col[j][t] -= d * col[i][t]; __syncthreads();
        }
        red[t] = col[j][t] * col[j][t]; __syncthreads();
        for (int s2 = 256; s2 > 0; s2 >>= 1) {
            if (t < s2) red[t] += red[t + s2];
            __syncthreads();
        }
        float nrm = sqrtf(fmaxf(red[0], 1e-20f)); __syncthreads();
        col[j][t] /= nrm; __syncthreads();
    }
    for (int j = 0; j < PK; ++j) V[t * PKP + j] = col[j][t];
}

__global__ void k_feats(const float* __restrict__ xc, const float* __restrict__ V,
                        float* __restrict__ feats)
{
    int i = blockIdx.x * blockDim.x + threadIdx.x;
    if (i >= NN) return;
    float acc[PK] = {};
    for (int k = 0; k < INC; ++k) {
        float v = xc[(size_t)i * INC + k];
#pragma unroll
        for (int j = 0; j < PK; ++j) acc[j] += v * V[k * PKP + j];
    }
#pragma unroll
    for (int j = 0; j < PK; ++j) feats[i * PKP + j] = acc[j];
}

// ---------------------------------------------------------------------------
// KMeans (Lloyd, 7 clusters, 10 dims)
// ---------------------------------------------------------------------------
__global__ void k_kmeans_seed(const float* __restrict__ feats, float* __restrict__ centers)
{
    int idx = threadIdx.x;
    if (idx >= KCL * PK) return;
    int c = idx / PK, j = idx % PK;
    centers[c * PKP + j] = feats[c * PKP + j];
}

__global__ void k_kmeans_assign(const float* __restrict__ feats,
                                const float* __restrict__ centers,
                                float* __restrict__ sums, float* __restrict__ cnt,
                                float* __restrict__ dist, int finalpass)
{
    int i = blockIdx.x * blockDim.x + threadIdx.x;
    if (i >= NN) return;
    float f[PK];
#pragma unroll
    for (int j = 0; j < PK; ++j) f[j] = feats[i * PKP + j];
    int best = 0; float bd = 1e30f;
    for (int c = 0; c < KCL; ++c) {
        float d = 0.f;
#pragma unroll
        for (int j = 0; j < PK; ++j) { float t = f[j] - centers[c * PKP + j]; d += t * t; }
        if (d < bd) { bd = d; best = c; }
    }
    if (!finalpass) {
#pragma unroll
        for (int j = 0; j < PK; ++j) atomicAdd(&sums[best * PKP + j], f[j]);
        atomicAdd(&cnt[best], 1.f);
    } else {
        dist[i] = sqrtf(bd);
    }
}

__global__ void k_kmeans_update(float* __restrict__ centers,
                                const float* __restrict__ sums,
                                const float* __restrict__ cnt)
{
    int c = blockIdx.x, j = threadIdx.x;
    if (j >= PK) return;
    float n = cnt[c];
    if (n > 0.f) centers[c * PKP + j] = sums[c * PKP + j] / fmaxf(n, 1.f);
}

// Median (4096 elems) via bitonic sort in LDS; write keep mask.
__global__ __launch_bounds__(1024)
void k_median_keep(const float* __restrict__ dist, float* __restrict__ keep)
{
    __shared__ float s[NN];
    __shared__ float med;
    for (int i = threadIdx.x; i < NN; i += 1024) s[i] = dist[i];
    __syncthreads();
    for (int k = 2; k <= NN; k <<= 1) {
        for (int j = k >> 1; j > 0; j >>= 1) {
            for (int t = threadIdx.x; t < NN; t += 1024) {
                int ixj = t ^ j;
                if (ixj > t) {
                    bool up = (t & k) == 0;
                    float a = s[t], b = s[ixj];
                    if ((a > b) == up) { s[t] = b; s[ixj] = a; }
                }
            }
            __syncthreads();
        }
    }
    if (threadIdx.x == 0) med = 0.5f * (s[NN / 2 - 1] + s[NN / 2]);
    __syncthreads();
    for (int i = threadIdx.x; i < NN; i += 1024)
        keep[i] = (dist[i] <= med) ? 1.f : 0.f;
}

// ---------------------------------------------------------------------------
// Cosine branch: segment softmax over out-edges of src
// ---------------------------------------------------------------------------
__device__ __forceinline__ unsigned f2ord(float f) {
    unsigned u = __float_as_uint(f);
    return (u & 0x80000000u) ? ~u : (u | 0x80000000u);
}
__device__ __forceinline__ float ord2f(unsigned u) {
    return __uint_as_float((u & 0x80000000u) ? (u & 0x7FFFFFFFu) : ~u);
}

__global__ __launch_bounds__(256)
void k_rownorm(const float* __restrict__ x, float* __restrict__ nx)
{
    int i = blockIdx.x;
    __shared__ float red[256];
    float s = 0.f;
    for (int c = threadIdx.x; c < INC; c += 256) {
        float v = x[(size_t)i * INC + c]; s += v * v;
    }
    red[threadIdx.x] = s; __syncthreads();
    for (int s2 = 128; s2 > 0; s2 >>= 1) {
        if (threadIdx.x < s2) red[threadIdx.x] += red[threadIdx.x + s2];
        __syncthreads();
    }
    float inv = 1.f / fmaxf(sqrtf(red[0]), 1e-12f);
    for (int c = threadIdx.x; c < INC; c += 256)
        nx[(size_t)i * INC + c] = x[(size_t)i * INC + c] * inv;
}

__global__ __launch_bounds__(256)
void k_sims(const float* __restrict__ nx, const int* __restrict__ src,
            const int* __restrict__ dst, float* __restrict__ sims, int E)
{
    int wid = (blockIdx.x * 256 + threadIdx.x) >> 5;   // wave32 per edge
    int lane = threadIdx.x & 31;
    if (wid >= E) return;
    int s = src[wid], d = dst[wid];
    float acc = 0.f;
    for (int c = lane; c < INC; c += 32)
        acc += nx[(size_t)s * INC + c] * nx[(size_t)d * INC + c];
    for (int o = 16; o > 0; o >>= 1) acc += __shfl_xor(acc, o, 32);
    if (lane == 0) sims[wid] = acc;
}

__global__ void k_smax(const float* __restrict__ sims, const int* __restrict__ src,
                       unsigned* __restrict__ smax, int E)
{
    int e = blockIdx.x * blockDim.x + threadIdx.x;
    if (e >= E) return;
    atomicMax(&smax[src[e]], f2ord(sims[e]));
}

__global__ void k_edge_exp(const float* __restrict__ sims, const unsigned* __restrict__ smax,
                           const int* __restrict__ src, float* __restrict__ ex,
                           float* __restrict__ den, float* __restrict__ deg, int E)
{
    int e = blockIdx.x * blockDim.x + threadIdx.x;
    if (e >= E) return;
    int s = src[e];
    float v = expf(sims[e] - ord2f(smax[s]));
    ex[e] = v;
    atomicAdd(&den[s], v);
    atomicAdd(&deg[s], 1.f);
}

__global__ void k_edge_w(const float* __restrict__ ex, const float* __restrict__ den,
                         const int* __restrict__ src, float* __restrict__ w,
                         float* __restrict__ wsum, int E)
{
    int e = blockIdx.x * blockDim.x + threadIdx.x;
    if (e >= E) return;
    int s = src[e];
    float v = ex[e] / fmaxf(den[s], 1e-12f);
    w[e] = v;
    atomicAdd(&wsum[s], v);
}

__global__ __launch_bounds__(256)
void k_cos_scatter(const float* __restrict__ w, const float* __restrict__ x,
                   const int* __restrict__ src, const int* __restrict__ dst,
                   float* __restrict__ num)
{
    int e = blockIdx.x;
    int s = src[e], d = dst[e];
    float we = w[e];
    for (int c = threadIdx.x; c < INC; c += 256)
        atomicAdd(&num[(size_t)s * INC + c], we * x[(size_t)d * INC + c]);
}

__global__ void k_cos_feats(const float* __restrict__ num, const float* __restrict__ wsum,
                            const float* __restrict__ deg, const float* __restrict__ x,
                            bf16* __restrict__ outb)
{
    int idx = blockIdx.x * blockDim.x + threadIdx.x;
    if (idx >= NN * INC) return;
    int i = idx >> 9;
    float v = (deg[i] > 0.f) ? num[idx] / fmaxf(wsum[i], 1e-12f) : x[idx];
    outb[idx] = (bf16)v;
}

// ---------------------------------------------------------------------------
// Message passing scatter + combined assembly + head
// ---------------------------------------------------------------------------
__global__ __launch_bounds__(256)
void k_scatter_h(const float* __restrict__ h, const int* __restrict__ src,
                 const int* __restrict__ dst, float* __restrict__ agg)
{
    int e = blockIdx.x;
    int s = src[e], d = dst[e];
    for (int c = threadIdx.x; c < HIDC; c += 256)
        atomicAdd(&agg[(size_t)d * HIDC + c], h[(size_t)s * HIDC + c]);
}

// which: 0 ego_enc(relu agg) | 1 cut(x*keep) | 2 cos_enc(relu agg) | 3 glob(plain)
__global__ void k_store_combined(const float* __restrict__ srcbuf,
                                 const float* __restrict__ x,
                                 const float* __restrict__ keep,
                                 bf16* __restrict__ comb, int which)
{
    int idx = blockIdx.x * blockDim.x + threadIdx.x;
    if (idx >= NN * 512) return;
    int i = idx >> 9, c = idx & 511;
    float v;
    if (which == 1) v = x[idx] * keep[i];
    else { v = srcbuf[idx]; if (which != 3) v = fmaxf(v, 0.f); }
    comb[(size_t)i * COMBC + which * 512 + c] = (bf16)v;
}

// W_fc[2048,40] f32 -> Wfct[48,2048] bf16 (transposed + zero-padded cols 40..47)
__global__ void k_pad_wfc_t(const float* __restrict__ W, bf16* __restrict__ Wt)
{
    int idx = blockIdx.x * blockDim.x + threadIdx.x;
    if (idx >= OUTP * COMBC) return;
    int c = idx / COMBC, r = idx % COMBC;
    Wt[idx] = (c < OUTC) ? (bf16)W[(size_t)r * OUTC + c] : (bf16)0.f;
}

__global__ void k_pad_bfc(const float* __restrict__ b, float* __restrict__ bp)
{
    int c = threadIdx.x;
    if (c < OUTP) bp[c] = (c < OUTC) ? b[c] : 0.f;
}

__global__ __launch_bounds__(64)
void k_logsoftmax(const float* __restrict__ logits, float* __restrict__ out)
{
    int i = blockIdx.x, t = threadIdx.x;
    __shared__ float red[64];
    float v = (t < OUTC) ? logits[(size_t)i * OUTP + t] : -1e30f;
    red[t] = v; __syncthreads();
    for (int s2 = 32; s2 > 0; s2 >>= 1) {
        if (t < s2) red[t] = fmaxf(red[t], red[t + s2]);
        __syncthreads();
    }
    float mx = red[0]; __syncthreads();
    float e = (t < OUTC) ? expf(v - mx) : 0.f;
    red[t] = e; __syncthreads();
    for (int s2 = 32; s2 > 0; s2 >>= 1) {
        if (t < s2) red[t] += red[t + s2];
        __syncthreads();
    }
    float lse = logf(red[0]);
    if (t < OUTC) out[(size_t)i * OUTC + t] = v - mx - lse;
}

// ---------------------------------------------------------------------------
// Host-side orchestration
// ---------------------------------------------------------------------------
static inline char* bump(char*& p, size_t bytes)
{
    char* r = p;
    p += (bytes + 255) & ~(size_t)255;
    return r;
}

extern "C" void kernel_launch(void* const* d_in, const int* in_sizes, int n_in,
                              void* d_out, int out_size, void* d_ws, size_t ws_size,
                              hipStream_t stream)
{
    const float* x      = (const float*)d_in[0];
    const float* W_ego  = (const float*)d_in[1];
    const float* b_ego  = (const float*)d_in[2];
    const float* W_cos  = (const float*)d_in[3];
    const float* b_cos  = (const float*)d_in[4];
    const float* W_glob = (const float*)d_in[5];
    const float* b_glob = (const float*)d_in[6];
    const float* W_fc   = (const float*)d_in[7];
    const float* b_fc   = (const float*)d_in[8];
    const int*   eidx   = (const int*)d_in[9];
    const int    E      = in_sizes[9] / 2;
    const int*   src    = eidx;
    const int*   dst    = eidx + E;

    // ---- workspace layout ----
    char* p = (char*)d_ws;
    unsigned* adj     = (unsigned*)bump(p, (size_t)NN * WORDS * 4);
    unsigned* reach   = (unsigned*)bump(p, (size_t)NN * WORDS * 4);
    bf16*  Mf         = (bf16*) bump(p, (size_t)NN * NN * 2);
    float* rowrecip   = (float*)bump(p, NN * 4);
    bf16*  xb         = (bf16*) bump(p, (size_t)NN * INC * 2);
    bf16*  xbT        = (bf16*) bump(p, (size_t)INC * NN * 2);
    float* mean       = (float*)bump(p, INC * 4);
    float* xc         = (float*)bump(p, (size_t)NN * INC * 4);
    bf16*  xcTb       = (bf16*) bump(p, (size_t)INC * NN * 2);
    float* cov        = (float*)bump(p, (size_t)INC * INC * 4);
    float* Vv         = (float*)bump(p, INC * PKP * 4);
    float* Uu         = (float*)bump(p, INC * PKP * 4);
    float* feats      = (float*)bump(p, (size_t)NN * PKP * 4);
    float* centers    = (float*)bump(p, KCL * PKP * 4);
    float* csums      = (float*)bump(p, KCL * PKP * 4);
    float* ccnt       = (float*)bump(p, KCL * 4);
    float* dist       = (float*)bump(p, NN * 4);
    float* keep       = (float*)bump(p, NN * 4);
    float* nx         = (float*)bump(p, (size_t)NN * INC * 4);
    float* sims       = (float*)bump(p, (size_t)E * 4);
    float* exbuf      = (float*)bump(p, (size_t)E * 4);
    float* wbuf       = (float*)bump(p, (size_t)E * 4);
    unsigned* smax    = (unsigned*)bump(p, NN * 4);
    float* den        = (float*)bump(p, NN * 4);
    float* wsum       = (float*)bump(p, NN * 4);
    float* deg        = (float*)bump(p, NN * 4);
    float* num        = (float*)bump(p, (size_t)NN * INC * 4);
    bf16*  cosb       = (bf16*) bump(p, (size_t)NN * INC * 2);
    bf16*  egob       = (bf16*) bump(p, (size_t)NN * INC * 2);
    float* hbuf       = (float*)bump(p, (size_t)NN * HIDC * 4);
    float* agg        = (float*)bump(p, (size_t)NN * HIDC * 4);
    float* gbuf       = (float*)bump(p, (size_t)NN * HIDC * 4);
    bf16*  WegoT      = (bf16*) bump(p, (size_t)INC * HIDC * 2);
    bf16*  WcosT      = (bf16*) bump(p, (size_t)INC * HIDC * 2);
    bf16*  WglobT     = (bf16*) bump(p, (size_t)INC * HIDC * 2);
    bf16*  WfcT       = (bf16*) bump(p, (size_t)OUTP * COMBC * 2);
    float* bfcp       = (float*)bump(p, OUTP * 4);
    bf16*  comb       = (bf16*) bump(p, (size_t)NN * COMBC * 2);
    float* logits     = (float*)bump(p, (size_t)NN * OUTP * 4);
    (void)ws_size; (void)n_in; (void)out_size;

    auto grid_of = [](int M, int Nc) { return dim3((Nc + 127) / 128, (M + 127) / 128); };

    // ============ weight / input conversions ============
    k_prep_x<<<(NN * INC + 255) / 256, 256, 0, stream>>>(x, xb, xbT);
    k_cvt_bf16_t<<<(INC * HIDC + 255) / 256, 256, 0, stream>>>(W_ego, WegoT, INC, HIDC);
    k_cvt_bf16_t<<<(INC * HIDC + 255) / 256, 256, 0, stream>>>(W_cos, WcosT, INC, HIDC);
    k_cvt_bf16_t<<<(INC * HIDC + 255) / 256, 256, 0, stream>>>(W_glob, WglobT, INC, HIDC);
    k_pad_wfc_t<<<(OUTP * COMBC + 255) / 256, 256, 0, stream>>>(W_fc, WfcT);
    k_pad_bfc<<<1, 64, 0, stream>>>(b_fc, bfcp);

    // ============ ego branch: bitset 2-hop reach ============
    hipMemsetAsync(adj, 0, (size_t)NN * WORDS * 4, stream);
    k_adjbits<<<(E + 255) / 256, 256, 0, stream>>>(src, dst, adj, E);
    k_reach_init<<<(NN * WORDS + 255) / 256, 256, 0, stream>>>(adj, reach);
    k_reach_2hop<<<E, 128, 0, stream>>>(src, dst, adj, reach);
    k_mf_expand<<<(NN * WORDS + 255) / 256, 256, 0, stream>>>(reach, Mf);
    k_rowrecip<<<(NN + 255) / 256, 256, 0, stream>>>(reach, rowrecip);
    // ego_feats (bf16) = (Mf @ x) * 1/rowsum  -- WMMA
    k_wmma_gemm_bf16<EPI_ROWSCALE | EPI_BF16OUT>
        <<<grid_of(NN, INC), 256, 0, stream>>>(Mf, xbT, nullptr, rowrecip, egob, NN, INC, NN);

    // ============ dominant branch: PCA subspace + kmeans ============
    k_colmean<<<INC, 256, 0, stream>>>(x, mean);
    k_center<<<(NN * INC + 255) / 256, 256, 0, stream>>>(x, mean, xc, xcTb);
    // cov = xc^T @ xc : A = xcT (512x4096), Bt = xcT (512x4096)  -- WMMA
    k_wmma_gemm_bf16<0>
        <<<grid_of(INC, INC), 256, 0, stream>>>(xcTb, xcTb, nullptr, nullptr, cov, INC, INC, NN);
    k_init_V<<<(INC * PKP + 255) / 256, 256, 0, stream>>>(Vv);
    for (int it = 0; it < 25; ++it) {
        k_cov_apply<<<2, 256, 0, stream>>>(cov, Vv, Uu);
        k_gram_schmidt<<<1, INC, 0, stream>>>(Uu, Vv);
    }
    k_feats<<<(NN + 255) / 256, 256, 0, stream>>>(xc, Vv, feats);
    k_kmeans_seed<<<1, 128, 0, stream>>>(feats, centers);
    for (int it = 0; it < 10; ++it) {
        hipMemsetAsync(csums, 0, KCL * PKP * 4, stream);
        hipMemsetAsync(ccnt, 0, KCL * 4, stream);
        k_kmeans_assign<<<(NN + 255) / 256, 256, 0, stream>>>(feats, centers, csums, ccnt, dist, 0);
        k_kmeans_update<<<KCL, 32, 0, stream>>>(centers, csums, ccnt);
    }
    k_kmeans_assign<<<(NN + 255) / 256, 256, 0, stream>>>(feats, centers, csums, ccnt, dist, 1);
    k_median_keep<<<1, 1024, 0, stream>>>(dist, keep);

    // ============ cosine branch ============
    k_rownorm<<<NN, 256, 0, stream>>>(x, nx);
    k_sims<<<(E * 32 + 255) / 256, 256, 0, stream>>>(nx, src, dst, sims, E);
    hipMemsetAsync(smax, 0, NN * 4, stream);   // ord(-NaN) floor
    hipMemsetAsync(den, 0, NN * 4, stream);
    hipMemsetAsync(wsum, 0, NN * 4, stream);
    hipMemsetAsync(deg, 0, NN * 4, stream);
    hipMemsetAsync(num, 0, (size_t)NN * INC * 4, stream);
    k_smax<<<(E + 255) / 256, 256, 0, stream>>>(sims, src, smax, E);
    k_edge_exp<<<(E + 255) / 256, 256, 0, stream>>>(sims, smax, src, exbuf, den, deg, E);
    k_edge_w<<<(E + 255) / 256, 256, 0, stream>>>(exbuf, den, src, wbuf, wsum, E);
    k_cos_scatter<<<E, 256, 0, stream>>>(wbuf, x, src, dst, num);
    k_cos_feats<<<(NN * INC + 255) / 256, 256, 0, stream>>>(num, wsum, deg, x, cosb);

    // ============ encoders (WMMA GEMM + edge scatter + relu) ============
    k_wmma_gemm_bf16<EPI_BIAS>
        <<<grid_of(NN, HIDC), 256, 0, stream>>>(xb, WglobT, b_glob, nullptr, gbuf, NN, HIDC, INC);
    // ego encoder
    k_wmma_gemm_bf16<EPI_BIAS>
        <<<grid_of(NN, HIDC), 256, 0, stream>>>(egob, WegoT, b_ego, nullptr, hbuf, NN, HIDC, INC);
    hipMemsetAsync(agg, 0, (size_t)NN * HIDC * 4, stream);
    k_scatter_h<<<E, 256, 0, stream>>>(hbuf, src, dst, agg);
    k_store_combined<<<(NN * 512 + 255) / 256, 256, 0, stream>>>(agg, x, keep, comb, 0);
    // cut_enc
    k_store_combined<<<(NN * 512 + 255) / 256, 256, 0, stream>>>(nullptr, x, keep, comb, 1);
    // cosine encoder
    k_wmma_gemm_bf16<EPI_BIAS>
        <<<grid_of(NN, HIDC), 256, 0, stream>>>(cosb, WcosT, b_cos, nullptr, hbuf, NN, HIDC, INC);
    hipMemsetAsync(agg, 0, (size_t)NN * HIDC * 4, stream);
    k_scatter_h<<<E, 256, 0, stream>>>(hbuf, src, dst, agg);
    k_store_combined<<<(NN * 512 + 255) / 256, 256, 0, stream>>>(agg, x, keep, comb, 2);
    // global feats into combined
    k_store_combined<<<(NN * 512 + 255) / 256, 256, 0, stream>>>(gbuf, x, keep, comb, 3);

    // ============ head: combined @ W_fc + b_fc, log_softmax ============
    k_wmma_gemm_bf16<EPI_BIAS>
        <<<grid_of(NN, OUTP), 256, 0, stream>>>(comb, WfcT, bfcp, nullptr, logits, NN, OUTP, COMBC);
    k_logsoftmax<<<NN, 64, 0, stream>>>(logits, (float*)d_out);
}
